// AttentionBlock_41695542509747
// MI455X (gfx1250) — matrix operand order repaired and checked by reading, not compiled, not run on hardware
//
#include <hip/hip_runtime.h>
#include <hip/hip_bf16.h>
#include <cstdint>

// ---------------------------------------------------------------------------
// AttentionBlock for MI455X (gfx1250): GroupNorm -> QKV -> flash attn -> proj
// Matmuls via v_wmma_f32_16x16x32_bf16 (wave32, fp32 accumulate).
// LDS staging via global_load_async_to_lds_b128 with double-buffered tiles:
// issue tile i+1, s_wait_asynccnt <= N (in-order completion) for tile i,
// overlap async DMA with WMMA math.
// ---------------------------------------------------------------------------

typedef __attribute__((ext_vector_type(16))) __bf16 v16bf;
typedef __attribute__((ext_vector_type(8)))  float  v8f;

union FragB { uint32_t u[8]; v16bf v; };

constexpr int   kC     = 512;
constexpr int   kHW    = 4096;
constexpr int   kM     = 2 * kHW;                 // B*HW = 8192
constexpr float kScale = 0.35355339059327373f;    // 64^-0.25

// CDNA5 async global->LDS copy, 16 bytes per lane. VDST = LDS byte offset
// (low 32 bits of the flat shared pointer are the wave-relative LDS address).
__device__ __forceinline__ void async_ld_b128(void* lds_ptr, const void* gptr) {
  uint32_t l = (uint32_t)(uintptr_t)lds_ptr;
  asm volatile("global_load_async_to_lds_b128 %0, %1, off"
               :: "v"(l), "v"(gptr) : "memory");
}
// Wait until at most `n` async ops remain outstanding (compile-time n).
__device__ __forceinline__ void async_wait_le(int n) {
  if (n == 0)      asm volatile("s_wait_asynccnt 0x0" ::: "memory");
  else if (n == 2) asm volatile("s_wait_asynccnt 0x2" ::: "memory");
  else if (n == 3) asm volatile("s_wait_asynccnt 0x3" ::: "memory");
  else             asm volatile("s_wait_asynccnt 0x0" ::: "memory");
}

// ---------------------------------------------------------------------------
// fp32 -> bf16 elementwise convert (weights)
// ---------------------------------------------------------------------------
__global__ void cvt_bf16(const float* __restrict__ in,
                         __hip_bfloat16* __restrict__ out, int n) {
  int i = blockIdx.x * blockDim.x + threadIdx.x;
  if (i < n) out[i] = __float2bfloat16(in[i]);
}

// ---------------------------------------------------------------------------
// GroupNorm over (C/G=16, HW=4096) per (b, g) + BCHW -> (B, HW, C) bf16
// grid = 64 blocks (b*32+g), 256 threads
// ---------------------------------------------------------------------------
__global__ __launch_bounds__(256)
void gn_kernel(const float* __restrict__ x,
               const float* __restrict__ gnw,
               const float* __restrict__ gnb,
               __hip_bfloat16* __restrict__ t) {
  __shared__ float r1[256];
  __shared__ float r2[256];
  __shared__ float tile[16][68];   // 16 channels x 64 hw, padded
  __shared__ float s_mean, s_rstd;

  const int tid = threadIdx.x;
  const int b = blockIdx.x >> 5;
  const int g = blockIdx.x & 31;
  const size_t base = ((size_t)b * kC + g * 16) * (size_t)kHW;

  // ---- phase 1: mean/var over 16*4096 = 65536 elements ----
  float s = 0.f, s2 = 0.f;
  for (int i = tid; i < 16 * kHW; i += 256) {
    float v = x[base + i];
    s += v; s2 += v * v;
  }
  r1[tid] = s; r2[tid] = s2;
  __syncthreads();
  for (int off = 128; off > 0; off >>= 1) {
    if (tid < off) { r1[tid] += r1[tid + off]; r2[tid] += r2[tid + off]; }
    __syncthreads();
  }
  if (tid == 0) {
    float mean = r1[0] * (1.f / 65536.f);
    float var  = r2[0] * (1.f / 65536.f) - mean * mean;
    s_mean = mean;
    s_rstd = rsqrtf(var + 1e-5f);
  }
  __syncthreads();
  const float mean = s_mean, rstd = s_rstd;

  // ---- phase 2: normalize + transpose to (B, HW, C) bf16, 64-hw tiles ----
  const int lr = tid >> 4;      // 0..15
  const int lc = tid & 15;      // 0..15
  const float gw = gnw[g * 16 + lc];
  const float gb = gnb[g * 16 + lc];
  for (int hw0 = 0; hw0 < kHW; hw0 += 64) {
    const float4 vin = *(const float4*)&x[base + (size_t)lr * kHW + hw0 + lc * 4];
    tile[lr][lc * 4 + 0] = vin.x;
    tile[lr][lc * 4 + 1] = vin.y;
    tile[lr][lc * 4 + 2] = vin.z;
    tile[lr][lc * 4 + 3] = vin.w;
    __syncthreads();
    #pragma unroll
    for (int j = 0; j < 4; ++j) {
      const int hw = hw0 + lr * 4 + j;
      float val = (tile[lc][lr * 4 + j] - mean) * rstd * gw + gb;
      t[((size_t)b * kHW + hw) * kC + g * 16 + lc] = __float2bfloat16(val);
    }
    __syncthreads();
  }
}

// ---------------------------------------------------------------------------
// bf16 GEMM: out[m,n] = f( sum_k A[m,k] * W[n,k] )
//   A: (8192, 512) bf16 row-major;  W: (512, 512) bf16 row-major
//   mode 0: outb[m*512+n] = bf16((acc + bias[n]) * scale)       [(B,HW,C)]
//   mode 1: outf[(b*512+n)*4096+hw] = acc + bias[n] + resid[..] [(B,C,HW)]
// grid = (64, 8), block 256 (8 waves); block tile 128x64, wave tile 32x32
// Double-buffered async LDS staging (3 async ops / thread / tile).
// ---------------------------------------------------------------------------
__global__ __launch_bounds__(256)
void gemm_bf16(const __hip_bfloat16* __restrict__ A,
               const __hip_bfloat16* __restrict__ W,
               const float* __restrict__ bias,
               float scale, int mode,
               __hip_bfloat16* __restrict__ outb,
               float* __restrict__ outf,
               const float* __restrict__ resid) {
  // stride 40 bf16 = 80 B: 16B-aligned rows for async b128; 20 words -> no
  // bank conflicts across the 16-lane half on fragment reads.
  __shared__ __align__(16) __hip_bfloat16 As[2][128][40];
  __shared__ __align__(16) __hip_bfloat16 Bs[2][64][40];

  const int tid   = threadIdx.x;
  const int lane  = tid & 31;
  const int wid   = tid >> 5;
  const int wm    = wid & 3;       // 4 m-subtiles of 32
  const int wn    = wid >> 2;      // 2 n-subtiles of 32
  const int lhalf = lane >> 4;     // 0/1
  const int l16   = lane & 15;
  const int m_base = blockIdx.x * 128;
  const int n_base = blockIdx.y * 64;

  const v8f zero = {0.f,0.f,0.f,0.f,0.f,0.f,0.f,0.f};
  v8f acc[2][2];
  for (int i = 0; i < 2; ++i) for (int j = 0; j < 2; ++j) acc[i][j] = zero;

  // stage one K-tile (A: 512 chunks of 16B -> 2/thread, B: 256 -> 1/thread)
  auto stage = [&](int buf, int kk) {
    #pragma unroll
    for (int j = 0; j < 2; ++j) {
      const int i = tid + j * 256;
      const int row = i >> 2, ch = i & 3;
      async_ld_b128(&As[buf][row][ch * 8],
                    A + (size_t)(m_base + row) * kC + kk + ch * 8);
    }
    const int row = tid >> 2, ch = tid & 3;
    async_ld_b128(&Bs[buf][row][ch * 8],
                  W + (size_t)(n_base + row) * kC + kk + ch * 8);
  };

  stage(0, 0);
  for (int it = 0; it < kC / 32; ++it) {
    const int kk  = it * 32;
    const int buf = it & 1;
    if (it + 1 < kC / 32) {
      stage(buf ^ 1, kk + 32);          // prefetch next tile into other buffer
      async_wait_le(3);                 // wait only for tile `it` (in-order)
    } else {
      async_wait_le(0);
    }
    if (kk + 64 < kC) {                 // L2 prefetch two tiles ahead
      __builtin_prefetch(A + (size_t)(m_base + (tid & 127)) * kC + kk + 64, 0, 3);
      __builtin_prefetch(W + (size_t)(n_base + (tid & 63)) * kC + kk + 64, 0, 3);
    }
    __syncthreads();

    FragB af[2], bfr[2];
    #pragma unroll
    for (int mt = 0; mt < 2; ++mt) {
      const int m = wm * 32 + mt * 16 + l16;
      #pragma unroll
      for (int v = 0; v < 8; ++v) {
        int kx = ((v & 3) * 2) + (lhalf ? 8 : 0) + ((v & 4) ? 16 : 0);
        af[mt].u[v] = *(const uint32_t*)&As[buf][m][kx];
      }
    }
    #pragma unroll
    for (int nt = 0; nt < 2; ++nt) {
      const int n = wn * 32 + nt * 16 + l16;
      #pragma unroll
      for (int v = 0; v < 8; ++v) {
        int kx = ((v & 3) * 2) + (lhalf ? 8 : 0) + ((v & 4) ? 16 : 0);
        bfr[nt].u[v] = *(const uint32_t*)&Bs[buf][n][kx];
      }
    }
    #pragma unroll
    for (int mt = 0; mt < 2; ++mt)
      #pragma unroll
      for (int nt = 0; nt < 2; ++nt)
        acc[mt][nt] = __builtin_amdgcn_wmma_f32_16x16x32_bf16(
            false, af[mt].v, false, bfr[nt].v, (short)0, acc[mt][nt], false, false);
    __syncthreads();
  }

  // epilogue (C/D layout: row = r + 8*half, col = lane%16)
  for (int mt = 0; mt < 2; ++mt) {
    for (int nt = 0; nt < 2; ++nt) {
      const int n = n_base + wn * 32 + nt * 16 + l16;
      const float bv = bias[n];
      #pragma unroll
      for (int r = 0; r < 8; ++r) {
        const int m = m_base + wm * 32 + mt * 16 + r + (lhalf ? 8 : 0);
        const float val = acc[mt][nt][r];
        if (mode == 0) {
          outb[(size_t)m * kC + n] = __float2bfloat16((val + bv) * scale);
        } else {
          const int bb = m >> 12, hw = m & 4095;
          const size_t idx = ((size_t)bb * kC + n) * (size_t)kHW + hw;
          // final output: written once, never re-read -> non-temporal
          float rv = __builtin_nontemporal_load(&resid[idx]);
          __builtin_nontemporal_store(val + bv + rv, &outf[idx]);
        }
      }
    }
  }
}

// ---------------------------------------------------------------------------
// Flash attention: one block per (b, head, 128 queries); 8 waves x 16 queries.
// Q/K/V in (B, HW, C) bf16, head slice = columns [head*64, head*64+64).
// Online softmax in fp32; P @ V accumulated with WMMA. Q staged once; K/V
// double-buffered with async b128 copies (2 async ops / thread / tile).
// grid = 2*8*32 = 512 blocks, 256 threads
// ---------------------------------------------------------------------------
__global__ __launch_bounds__(256)
void attn_kernel(const __hip_bfloat16* __restrict__ Q,
                 const __hip_bfloat16* __restrict__ K,
                 const __hip_bfloat16* __restrict__ V,
                 __hip_bfloat16* __restrict__ O) {
  // stride 72 bf16 = 144 B: 16B-aligned, 36 words -> conflict-free frag reads
  __shared__ __align__(16) __hip_bfloat16 Qs[128][72];       // 128 q x 64 d
  __shared__ __align__(16) __hip_bfloat16 Ks[2][32][72];     // 32 keys x 64 d
  __shared__ __align__(16) __hip_bfloat16 Vs[2][32][72];     // 32 keys x 64 d
  __shared__ __align__(16) __hip_bfloat16 Ps[8][16][34];     // per-wave P tile

  const int tid   = threadIdx.x;
  const int lane  = tid & 31;
  const int w     = tid >> 5;
  const int lhalf = lane >> 4;
  const int l16   = lane & 15;

  const int qb   = blockIdx.x & 31;
  const int bh   = blockIdx.x >> 5;
  const int head = bh & 7;
  const int b    = bh >> 3;
  const size_t base = (size_t)b * kHW * kC + head * 64;
  const int q0 = qb * 128;

  auto stage_kv = [&](int buf, int kb) {
    const int row = tid >> 3, ch = tid & 7;
    async_ld_b128(&Ks[buf][row][ch * 8],
                  K + base + (size_t)(kb + row) * kC + ch * 8);
    async_ld_b128(&Vs[buf][row][ch * 8],
                  V + base + (size_t)(kb + row) * kC + ch * 8);
  };

  // ---- async stage Q block once: 128 rows x 128B = 1024 x 16B chunks ----
  #pragma unroll
  for (int j = 0; j < 4; ++j) {
    const int i = tid + j * 256;
    const int row = i >> 3, ch = i & 7;
    async_ld_b128(&Qs[row][ch * 8],
                  Q + base + (size_t)(q0 + row) * kC + ch * 8);
  }
  stage_kv(0, 0);          // first K/V tile rides behind the Q copy
  async_wait_le(2);        // Q done (in-order); K/V tile 0 may still be going
  __syncthreads();

  // per-wave Q fragments for d=[0,32) and [32,64)
  FragB qf[2];
  #pragma unroll
  for (int ks = 0; ks < 2; ++ks)
    #pragma unroll
    for (int v = 0; v < 8; ++v) {
      int kx = ks * 32 + (v & 3) * 2 + (lhalf ? 8 : 0) + ((v & 4) ? 16 : 0);
      qf[ks].u[v] = *(const uint32_t*)&Qs[w * 16 + l16][kx];
    }

  const v8f zero = {0.f,0.f,0.f,0.f,0.f,0.f,0.f,0.f};
  v8f oacc[4];
  #pragma unroll
  for (int dt = 0; dt < 4; ++dt) oacc[dt] = zero;
  float mi[8], li[8];
  #pragma unroll
  for (int r = 0; r < 8; ++r) { mi[r] = -1e30f; li[r] = 0.f; }

  for (int it = 0; it < kHW / 32; ++it) {
    const int kb  = it * 32;
    const int buf = it & 1;
    if (it + 1 < kHW / 32) {
      stage_kv(buf ^ 1, kb + 32);   // prefetch next K/V tile
      async_wait_le(2);             // wait only for tile `it`
    } else {
      async_wait_le(0);
    }
    if (kb + 64 < kHW) {            // L2 prefetch two tiles ahead
      __builtin_prefetch(K + base + (size_t)(kb + 64 + (tid & 31)) * kC, 0, 3);
      __builtin_prefetch(V + base + (size_t)(kb + 64 + (tid & 31)) * kC, 0, 3);
    }
    __syncthreads();

    // S = q . k^T  (16 queries x 32 keys per wave)
    v8f sacc[2]; sacc[0] = zero; sacc[1] = zero;
    #pragma unroll
    for (int nt = 0; nt < 2; ++nt) {
      #pragma unroll
      for (int ks = 0; ks < 2; ++ks) {
        FragB kfB;
        #pragma unroll
        for (int v = 0; v < 8; ++v) {
          int kx = ks * 32 + (v & 3) * 2 + (lhalf ? 8 : 0) + ((v & 4) ? 16 : 0);
          kfB.u[v] = *(const uint32_t*)&Ks[buf][nt * 16 + l16][kx];
        }
        sacc[nt] = __builtin_amdgcn_wmma_f32_16x16x32_bf16(
            false, qf[ks].v, false, kfB.v, (short)0, sacc[nt], false, false);
      }
    }

    // online softmax: rows live at (r, half); reduce across 16-lane halves
    #pragma unroll
    for (int r = 0; r < 8; ++r) {
      float s0 = sacc[0][r], s1 = sacc[1][r];
      float mx = fmaxf(s0, s1);
      mx = fmaxf(mx, __shfl_xor(mx, 1));
      mx = fmaxf(mx, __shfl_xor(mx, 2));
      mx = fmaxf(mx, __shfl_xor(mx, 4));
      mx = fmaxf(mx, __shfl_xor(mx, 8));
      float mnew = fmaxf(mi[r], mx);
      float corr = __expf(mi[r] - mnew);
      float p0 = __expf(s0 - mnew);
      float p1 = __expf(s1 - mnew);
      float rs = p0 + p1;
      rs += __shfl_xor(rs, 1);
      rs += __shfl_xor(rs, 2);
      rs += __shfl_xor(rs, 4);
      rs += __shfl_xor(rs, 8);
      li[r] = li[r] * corr + rs;
      mi[r] = mnew;
      const int mrow = r + (lhalf ? 8 : 0);
      Ps[w][mrow][l16]      = __float2bfloat16(p0);
      Ps[w][mrow][16 + l16] = __float2bfloat16(p1);
      #pragma unroll
      for (int dt = 0; dt < 4; ++dt) oacc[dt][r] *= corr;
    }

    // reload P as an A-fragment (per-wave LDS region, no barrier needed)
    FragB pf;
    #pragma unroll
    for (int v = 0; v < 8; ++v) {
      int kx = (v & 3) * 2 + (lhalf ? 8 : 0) + ((v & 4) ? 16 : 0);
      pf.u[v] = *(const uint32_t*)&Ps[w][l16][kx];
    }
    // O += P @ V : B-fragment packed from two u16 loads (adjacent keys)
    #pragma unroll
    for (int dt = 0; dt < 4; ++dt) {
      FragB vf;
      #pragma unroll
      for (int v = 0; v < 8; ++v) {
        const int k0 = (v & 3) * 2 + (lhalf ? 8 : 0) + ((v & 4) ? 16 : 0);
        uint32_t lo = *(const uint16_t*)&Vs[buf][k0][dt * 16 + l16];
        uint32_t hi = *(const uint16_t*)&Vs[buf][k0 + 1][dt * 16 + l16];
        vf.u[v] = lo | (hi << 16);
      }
      oacc[dt] = __builtin_amdgcn_wmma_f32_16x16x32_bf16(
          false, pf.v, false, vf.v, (short)0, oacc[dt], false, false);
    }
    __syncthreads();
  }

  // write O (divide by softmax denominator)
  #pragma unroll
  for (int dt = 0; dt < 4; ++dt) {
    #pragma unroll
    for (int r = 0; r < 8; ++r) {
      const int m = q0 + w * 16 + r + (lhalf ? 8 : 0);
      const int d = dt * 16 + l16;
      const float val = oacc[dt][r] / li[r];
      O[base + (size_t)m * kC + d] = __float2bfloat16(val);
    }
  }
}

// ---------------------------------------------------------------------------
extern "C" void kernel_launch(void* const* d_in, const int* in_sizes, int n_in,
                              void* d_out, int out_size, void* d_ws, size_t ws_size,
                              hipStream_t stream) {
  (void)in_sizes; (void)n_in; (void)out_size; (void)ws_size;
  const float* x   = (const float*)d_in[0];
  const float* gnw = (const float*)d_in[1];
  const float* gnb = (const float*)d_in[2];
  const float* wq  = (const float*)d_in[3];
  const float* bq  = (const float*)d_in[4];
  const float* wk  = (const float*)d_in[5];
  const float* bk  = (const float*)d_in[6];
  const float* wv  = (const float*)d_in[7];
  const float* bv  = (const float*)d_in[8];
  const float* wp  = (const float*)d_in[9];
  const float* bp  = (const float*)d_in[10];
  float* out = (float*)d_out;

  // workspace layout (~42 MB total)
  char* ws = (char*)d_ws;
  size_t off = 0;
  auto take = [&](size_t bytes) {
    void* p = ws + off;
    off += (bytes + 255) & ~(size_t)255;
    return p;
  };
  const size_t wbytes = (size_t)kC * kC * sizeof(__hip_bfloat16);
  const size_t tbytes = (size_t)kM * kC * sizeof(__hip_bfloat16);
  __hip_bfloat16* wq_bf = (__hip_bfloat16*)take(wbytes);
  __hip_bfloat16* wk_bf = (__hip_bfloat16*)take(wbytes);
  __hip_bfloat16* wv_bf = (__hip_bfloat16*)take(wbytes);
  __hip_bfloat16* wp_bf = (__hip_bfloat16*)take(wbytes);
  __hip_bfloat16* tbuf  = (__hip_bfloat16*)take(tbytes);
  __hip_bfloat16* qbuf  = (__hip_bfloat16*)take(tbytes);
  __hip_bfloat16* kbuf  = (__hip_bfloat16*)take(tbytes);
  __hip_bfloat16* vbuf  = (__hip_bfloat16*)take(tbytes);
  __hip_bfloat16* obuf  = (__hip_bfloat16*)take(tbytes);

  const int wn = kC * kC;
  const int cvtBlocks = (wn + 255) / 256;
  cvt_bf16<<<cvtBlocks, 256, 0, stream>>>(wq, wq_bf, wn);
  cvt_bf16<<<cvtBlocks, 256, 0, stream>>>(wk, wk_bf, wn);
  cvt_bf16<<<cvtBlocks, 256, 0, stream>>>(wv, wv_bf, wn);
  cvt_bf16<<<cvtBlocks, 256, 0, stream>>>(wp, wp_bf, wn);

  gn_kernel<<<64, 256, 0, stream>>>(x, gnw, gnb, tbuf);

  dim3 ggrid(kM / 128, kC / 64);  // 64 x 8
  gemm_bf16<<<ggrid, 256, 0, stream>>>(tbuf, wq_bf, bq, kScale, 0, qbuf, nullptr, nullptr);
  gemm_bf16<<<ggrid, 256, 0, stream>>>(tbuf, wk_bf, bk, kScale, 0, kbuf, nullptr, nullptr);
  gemm_bf16<<<ggrid, 256, 0, stream>>>(tbuf, wv_bf, bv, 1.0f,   0, vbuf, nullptr, nullptr);

  attn_kernel<<<512, 256, 0, stream>>>(qbuf, kbuf, vbuf, obuf);

  gemm_bf16<<<ggrid, 256, 0, stream>>>(obuf, wp_bf, bp, 1.0f, 1, nullptr, out, x);
}
